// TransformerEncoder_33629593928312
// MI455X (gfx1250) — matrix operand order repaired and checked by reading, not compiled, not run on hardware
//
#include <hip/hip_runtime.h>

typedef _Float16 f16;
typedef __attribute__((ext_vector_type(16))) _Float16 v16h;
typedef __attribute__((ext_vector_type(4)))  _Float16 v4h;
typedef __attribute__((ext_vector_type(8)))  float    v8f;

#define HD     1024
#define S1T    150
#define QROWS  160          // padded token rows per batch in qkv buffer
#define NB     128
#define MTOK   (S1T * NB)   // 19200 tokens

// ---- WMMA fragment index maps (CDNA5 ISA 7.12.2, 16-bit operands, wave32) ----
__device__ __forceinline__ int ka_idx(int h, int lane) {
    return (h & 7) + ((h >> 3) << 4) + ((lane >> 4) << 3);
}
__device__ __forceinline__ int kb_idx(int h, int lane) {
    return (h & 7) + ((h >> 3) << 4) + ((lane >> 4) << 3);
}

// ---- CDNA5 async global->LDS copy (ASYNCcnt-tracked, no VGPR round-trip) ----
__device__ __forceinline__ void async_ld_b128(void* lds, const void* gp) {
    uint32_t l32 = (uint32_t)(uintptr_t)lds;      // flat LDS addr truncates to LDS offset
    uint64_t g64 = (uint64_t)(uintptr_t)gp;
    asm volatile("global_load_async_to_lds_b128 %0, %1, off"
                 :: "v"(l32), "v"(g64) : "memory");
}
__device__ __forceinline__ void wait_async0() {
    asm volatile("s_wait_asynccnt 0" ::: "memory");
}

// ---------------------------------------------------------------- weights cvt
__global__ void cvt_f32_f16_kernel(const float* __restrict__ src,
                                   f16* __restrict__ dst, int n) {
    int i = blockIdx.x * blockDim.x + threadIdx.x;
    if (i < n) dst[i] = (f16)src[i];
}

// ------------------------------------------------------------- PE table (once)
__global__ void pe_kernel(float* __restrict__ pe) {
    int i = blockIdx.x * blockDim.x + threadIdx.x;
    if (i < S1T * HD) {
        int s = i >> 10, c = i & (HD - 1);
        int i2 = c & ~1;
        float freq = __expf((float)i2 * (-9.210340371976184f / (float)HD));
        float ang = 6.283185307179586f * (float)s * freq;
        pe[i] = (c & 1) ? cosf(ang) : sinf(ang);
    }
}

// -------------------------------------------- zero the qkv padding rows (once)
__global__ void zero_pad_kernel(f16* __restrict__ qkv) {
    int i = blockIdx.x * blockDim.x + threadIdx.x;
    int total = NB * (QROWS - S1T) * 3 * HD / 8;
    if (i < total) {
        int perb = (QROWS - S1T) * 3 * HD / 8;
        int b = i / perb, off = i - b * perb;
        uint4 zz = {0, 0, 0, 0};
        *(uint4*)(qkv + ((size_t)b * QROWS + S1T) * (3 * HD) + (size_t)off * 8) = zz;
    }
}

// -------------------------------------------------------------- embed + LN
__global__ __launch_bounds__(256)
void embed_ln_kernel(const float* __restrict__ z, const float* __restrict__ cls,
                     const float* __restrict__ pe,
                     const float* __restrict__ g, const float* __restrict__ bb,
                     float* __restrict__ o32, f16* __restrict__ o16) {
    int m = blockIdx.x;                  // token index = b*S1 + s
    int b = m / S1T, s = m - b * S1T;
    int tid = threadIdx.x;
    int lane = tid & 31, wid = tid >> 5;

    const float4* srcp = (s == 0) ? (const float4*)cls
                                  : (const float4*)(z + ((size_t)(s - 1) * NB + b) * HD);
    float4 sv = srcp[tid];
    float4 pv = ((const float4*)(pe + (size_t)s * HD))[tid];
    float v[4] = {sv.x + pv.x, sv.y + pv.y, sv.z + pv.z, sv.w + pv.w};
    float sum = v[0] + v[1] + v[2] + v[3];
    float sum2 = v[0]*v[0] + v[1]*v[1] + v[2]*v[2] + v[3]*v[3];
#pragma unroll
    for (int off = 16; off > 0; off >>= 1) {
        sum  += __shfl_xor(sum, off);
        sum2 += __shfl_xor(sum2, off);
    }
    __shared__ float sb1[8], sb2[8];
    if (lane == 0) { sb1[wid] = sum; sb2[wid] = sum2; }
    __syncthreads();
    float S = 0.f, S2 = 0.f;
#pragma unroll
    for (int i = 0; i < 8; ++i) { S += sb1[i]; S2 += sb2[i]; }
    float mean = S * (1.0f / HD);
    float var  = S2 * (1.0f / HD) - mean * mean;
    float inv  = rsqrtf(var + 1e-5f);
    float4 gv = ((const float4*)g)[tid];
    float4 bv = ((const float4*)bb)[tid];
    float y0 = (v[0]-mean)*inv*gv.x + bv.x;
    float y1 = (v[1]-mean)*inv*gv.y + bv.y;
    float y2 = (v[2]-mean)*inv*gv.z + bv.z;
    float y3 = (v[3]-mean)*inv*gv.w + bv.w;
    ((float4*)(o32 + (size_t)m * HD))[tid] = make_float4(y0, y1, y2, y3);
    v4h hv = {(f16)y0, (f16)y1, (f16)y2, (f16)y3};
    *(v4h*)(o16 + (size_t)m * HD + tid * 4) = hv;
}

// -------------------------------------------------------------------- LN
__global__ __launch_bounds__(256)
void ln_kernel(const float* __restrict__ in, const float* __restrict__ g,
               const float* __restrict__ bb, float* __restrict__ o32,
               f16* __restrict__ o16) {
    int m = blockIdx.x;
    int tid = threadIdx.x, lane = tid & 31, wid = tid >> 5;
    const float4 v4 = ((const float4*)(in + (size_t)m * HD))[tid];
    float v[4] = {v4.x, v4.y, v4.z, v4.w};
    float sum = v[0] + v[1] + v[2] + v[3];
    float sum2 = v[0]*v[0] + v[1]*v[1] + v[2]*v[2] + v[3]*v[3];
#pragma unroll
    for (int off = 16; off > 0; off >>= 1) {
        sum  += __shfl_xor(sum, off);
        sum2 += __shfl_xor(sum2, off);
    }
    __shared__ float sb1[8], sb2[8];
    if (lane == 0) { sb1[wid] = sum; sb2[wid] = sum2; }
    __syncthreads();
    float S = 0.f, S2 = 0.f;
#pragma unroll
    for (int i = 0; i < 8; ++i) { S += sb1[i]; S2 += sb2[i]; }
    float mean = S * (1.0f / HD);
    float var  = S2 * (1.0f / HD) - mean * mean;
    float inv  = rsqrtf(var + 1e-5f);
    float4 gv = ((const float4*)g)[tid];
    float4 bv = ((const float4*)bb)[tid];
    float y0 = (v[0]-mean)*inv*gv.x + bv.x;
    float y1 = (v[1]-mean)*inv*gv.y + bv.y;
    float y2 = (v[2]-mean)*inv*gv.z + bv.z;
    float y3 = (v[3]-mean)*inv*gv.w + bv.w;
    ((float4*)(o32 + (size_t)m * HD))[tid] = make_float4(y0, y1, y2, y3);
    v4h hv = {(f16)y0, (f16)y1, (f16)y2, (f16)y3};
    *(v4h*)(o16 + (size_t)m * HD + tid * 4) = hv;
}

// ------------------------------------------------------------ tiled WMMA GEMM
// 64x64 block tile, 8 waves, double-buffered LDS with async global->LDS DMA:
// chunk k+1 streams into LDS while chunk k runs v_wmma.
#define ASTR 40
#define BSTR 72
template <int MODE>
__global__ __launch_bounds__(256)
void gemm_f16_kernel(const f16* __restrict__ A, const f16* __restrict__ Bw,
                     const float* __restrict__ bias, const float* __restrict__ res,
                     f16* __restrict__ outh, float* __restrict__ outf,
                     int M, int N, int K) {
    __shared__ __align__(16) f16 As[2 * 64 * ASTR];
    __shared__ __align__(16) f16 Bs[2 * 32 * BSTR];
    int tid = threadIdx.x, lane = tid & 31, wid = tid >> 5;
    int wm = wid & 3, wn = wid >> 2;
    int bm0 = blockIdx.y * 64, bn0 = blockIdx.x * 64;

    v8f c0 = {}; v8f c1 = {};
    int arow = tid >> 2, acg = (tid & 3) * 8;   // A stage: 64x32, b128/thread
    int brow = tid >> 3, bcg = (tid & 7) * 8;   // B stage: 32x64, b128/thread

    auto stage = [&](int buf, int kc) {
        async_ld_b128(As + buf * 64 * ASTR + arow * ASTR + acg,
                      A + (size_t)(bm0 + arow) * K + kc + acg);
        async_ld_b128(Bs + buf * 32 * BSTR + brow * BSTR + bcg,
                      Bw + (size_t)(kc + brow) * N + bn0 + bcg);
    };

    stage(0, 0);
    wait_async0();
    __syncthreads();
    int buf = 0;
    for (int kc = 0; kc < K; kc += 32) {
        if (kc + 32 < K) stage(buf ^ 1, kc + 32);
        const f16* ab = As + buf * 64 * ASTR;
        const f16* bbf = Bs + buf * 32 * BSTR;
        v16h a, b0, b1;
        int mr = wm * 16 + (lane & 15);
        int nc = wn * 32 + (lane & 15);
#pragma unroll
        for (int h = 0; h < 16; ++h) {
            a[h] = ab[mr * ASTR + ka_idx(h, lane)];
            int kb = kb_idx(h, lane);
            b0[h] = bbf[kb * BSTR + nc];
            b1[h] = bbf[kb * BSTR + nc + 16];
        }
        c0 = __builtin_amdgcn_wmma_f32_16x16x32_f16(false, a, false, b0,
                                                    (short)0, c0, false, false);
        c1 = __builtin_amdgcn_wmma_f32_16x16x32_f16(false, a, false, b1,
                                                    (short)0, c1, false, false);
        wait_async0();
        __syncthreads();
        buf ^= 1;
    }

    // epilogue: C layout = VGPR r -> M = r (+8 for lanes 16-31), N = lane&15
    int gm0 = bm0 + wm * 16 + ((lane >> 4) << 3);
    int gn0 = bn0 + wn * 32 + (lane & 15);
#pragma unroll
    for (int r = 0; r < 8; ++r) {
        int gm = gm0 + r;
#pragma unroll
        for (int t = 0; t < 2; ++t) {
            int gn = gn0 + t * 16;
            float val = (t ? c1[r] : c0[r]) + bias[gn];
            size_t idx = (size_t)gm * N + gn;
            if (MODE == 0) {
                int b = gm / S1T, s = gm - b * S1T;
                outh[((size_t)b * QROWS + s) * (size_t)N + gn] = (f16)val;
            } else if (MODE == 1) {
                outf[idx] = val + res[idx];
            } else if (MODE == 2) {
                float ge = 0.5f * val * (1.0f + erff(val * 0.7071067811865476f));
                outh[idx] = (f16)ge;
            } else {
                float y = val + res[idx];
                int b = gm / S1T, s = gm - b * S1T;
                outf[((size_t)s * NB + b) * (size_t)N + gn] = y;  // (S1,B,H)
            }
        }
    }
}

// ----------------------------------------------------------------- attention
// Padded qkv (160 rows/batch, tail zeroed). One block per (16-query tile,
// batch), 10 waves. Phase A double-buffered with async DMA.
#define KSTR 72
#define QKBUF (16 * KSTR + 160 * KSTR)   // halves per phase-A buffer (12672)
__global__ __launch_bounds__(320)
void attn_kernel(const f16* __restrict__ qkv, f16* __restrict__ out) {
    __shared__ __align__(16) char smemA[2 * QKBUF * 2];   // 50688 B
    __shared__ __align__(16) f16 ps[16 * 168];            // probs
    float* ssm = (float*)smemA;               // 16 x 160 scores (after phase A)
    f16*   vs  = (f16*)smemA;                 // 32 x 512 V chunk (phase B)

    int tid = threadIdx.x, lane = tid & 31;
    int w = __builtin_amdgcn_readfirstlane(tid >> 5);   // wave id 0..9 (scalar)
    int qt = blockIdx.x;                                // query tile 0..9
    int bidx = blockIdx.y;                              // batch 0..127
    const f16* qkvb = qkv + (size_t)bidx * QROWS * (3 * HD);

    // ---------------- phase A: scores = Q K^T (double-buffered) ----------------
    auto stageA = [&](int buf, int hc) {
        f16* qsb = (f16*)smemA + buf * QKBUF;
        f16* ksb = qsb + 16 * KSTR;
        if (tid < 128) {      // qs: 16 x 64, one b128 per thread
            int r = tid >> 3, cg = (tid & 7) * 8;
            async_ld_b128(qsb + r * KSTR + cg,
                          qkvb + (size_t)(qt * 16 + r) * (3 * HD) + hc + cg);
        }
        {                     // ks: 160 x 64, four b128 per thread
            int r = tid >> 1, cg = (tid & 1) * 32;
            const f16* src = qkvb + (size_t)r * (3 * HD) + HD + hc + cg;
            f16* dst = ksb + r * KSTR + cg;
#pragma unroll
            for (int i = 0; i < 4; ++i)
                async_ld_b128(dst + i * 8, src + i * 8);
        }
    };

    v8f sc = {};
    stageA(0, 0);
    wait_async0();
    __syncthreads();
    int bufA = 0;
    for (int hc = 0; hc < HD; hc += 64) {
        if (hc + 64 < HD) stageA(bufA ^ 1, hc + 64);
        const f16* qsb = (const f16*)smemA + bufA * QKBUF;
        const f16* ksb = qsb + 16 * KSTR;
#pragma unroll
        for (int sub = 0; sub < 2; ++sub) {
            v16h aq, bk;
            int qr = lane & 15;
            int kr = w * 16 + (lane & 15);
#pragma unroll
            for (int h = 0; h < 16; ++h) {
                aq[h] = qsb[qr * KSTR + sub * 32 + ka_idx(h, lane)];
                bk[h] = ksb[kr * KSTR + sub * 32 + kb_idx(h, lane)];
            }
            sc = __builtin_amdgcn_wmma_f32_16x16x32_f16(false, aq, false, bk,
                                                        (short)0, sc, false, false);
        }
        wait_async0();
        __syncthreads();
        bufA ^= 1;
    }

    // scale + band mask, write scores to LDS (aliases phase-A buffers)
    {
        int kcol = lane & 15;
        int kk = w * 16 + kcol;
        int qr0 = (lane >> 4) << 3;
#pragma unroll
        for (int r = 0; r < 8; ++r) {
            int qr = qr0 + r;
            int qq = qt * 16 + qr;
            float sval = sc[r] * 0.03125f;   // 1/sqrt(1024)
            bool ok;
            if (qq == 0) ok = (kk < S1T);
            else {
                int lo = qq - 20; if (lo < 0) lo = 0;
                ok = (qq < S1T) && (kk < S1T) && (kk >= lo) && (kk < qq + 20);
            }
            ssm[qr * 160 + kk] = ok ? sval : -1e30f;
        }
    }
    __syncthreads();

    // ---------------- softmax (rows spread over waves) ----------------
    for (int r = w; r < 16; r += 10) {
        float vv[5]; float mx = -3.0e38f;
#pragma unroll
        for (int j = 0; j < 5; ++j) {
            vv[j] = ssm[r * 160 + lane + 32 * j];
            mx = fmaxf(mx, vv[j]);
        }
#pragma unroll
        for (int off = 16; off > 0; off >>= 1) mx = fmaxf(mx, __shfl_xor(mx, off));
        float ee[5]; float sum = 0.f;
#pragma unroll
        for (int j = 0; j < 5; ++j) { ee[j] = __expf(vv[j] - mx); sum += ee[j]; }
#pragma unroll
        for (int off = 16; off > 0; off >>= 1) sum += __shfl_xor(sum, off);
        float inv = 1.0f / sum;
#pragma unroll
        for (int j = 0; j < 5; ++j)
            ps[r * 168 + lane + 32 * j] = (f16)(ee[j] * inv);
    }
    __syncthreads();

    // ---------------- phase B: out = P @ V ----------------
    v8f zf = {0.f, 0.f, 0.f, 0.f, 0.f, 0.f, 0.f, 0.f};
    v8f acc[8];
#pragma unroll
    for (int i = 0; i < 8; ++i) acc[i] = zf;

    for (int kc2 = 0; kc2 < 160; kc2 += 32) {
        v16h pa;
        {
            int qr = lane & 15;
#pragma unroll
            for (int h = 0; h < 16; ++h)
                pa[h] = ps[qr * 168 + kc2 + ka_idx(h, lane)];
        }
#pragma unroll
        for (int hb = 0; hb < 2; ++hb) {         // V column halves (512 each)
            __syncthreads();                     // protect vs overwrite
            for (int i8 = tid * 8; i8 < 32 * 512; i8 += 320 * 8) {
                int rr = i8 >> 9, cc = i8 & 511;
                async_ld_b128(vs + i8,
                              qkvb + (size_t)(kc2 + rr) * (3 * HD) +
                              2 * HD + hb * 512 + cc);
            }
            wait_async0();
            __syncthreads();
#pragma unroll
            for (int i = 0; i < 4; ++i) {
                int t = w + 10 * i;              // scalar (w is readfirstlane)
                if (t < 32) {
                    v16h vb;
                    int ncol = t * 16 + (lane & 15);
#pragma unroll
                    for (int h = 0; h < 16; ++h)
                        vb[h] = vs[kb_idx(h, lane) * 512 + ncol];
                    acc[hb * 4 + i] = __builtin_amdgcn_wmma_f32_16x16x32_f16(
                        false, pa, false, vb, (short)0, acc[hb * 4 + i],
                        false, false);
                }
            }
        }
    }

    // store attention output (f16, unpadded 150-row layout for O-proj GEMM)
    {
        int qr0 = (lane >> 4) << 3;
        int ncl = lane & 15;
#pragma unroll
        for (int hb = 0; hb < 2; ++hb)
#pragma unroll
            for (int i = 0; i < 4; ++i) {
                int t = w + 10 * i;
                if (t < 32) {
                    int gh = (hb * 32 + t) * 16 + ncl;
#pragma unroll
                    for (int r = 0; r < 8; ++r) {
                        int qq = qt * 16 + qr0 + r;
                        if (qq < S1T)
                            out[((size_t)bidx * S1T + qq) * HD + gh] =
                                (f16)acc[hb * 4 + i][r];
                    }
                }
            }
    }
}

// ---------------------------------------------------------------- launcher
extern "C" void kernel_launch(void* const* d_in, const int* in_sizes, int n_in,
                              void* d_out, int out_size, void* d_ws, size_t ws_size,
                              hipStream_t stream) {
    (void)in_sizes; (void)n_in; (void)out_size; (void)ws_size;
    const float* z     = (const float*)d_in[0];
    const float* cls   = (const float*)d_in[1];
    const float* ln_g  = (const float*)d_in[2];
    const float* ln_b  = (const float*)d_in[3];
    const float* w_qkv = (const float*)d_in[4];
    const float* b_qkv = (const float*)d_in[5];
    const float* w_o   = (const float*)d_in[6];
    const float* b_o   = (const float*)d_in[7];
    const float* w_m1  = (const float*)d_in[8];
    const float* b_m1  = (const float*)d_in[9];
    const float* w_m2  = (const float*)d_in[10];
    const float* b_m2  = (const float*)d_in[11];
    float* out = (float*)d_out;

    char* p = (char*)d_ws;
    auto take = [&](size_t bytes) -> char* {
        char* r = p; p += (bytes + 255) & ~(size_t)255; return r;
    };
    f16*   wqkv16 = (f16*)take((size_t)3 * HD * HD * 2);
    f16*   wo16   = (f16*)take((size_t)HD * HD * 2);
    f16*   wm116  = (f16*)take((size_t)HD * HD * 2);
    f16*   wm216  = (f16*)take((size_t)HD * HD * 2);
    float* petab  = (float*)take((size_t)S1T * HD * 4);
    float* x32    = (float*)take((size_t)MTOK * HD * 4);         // x / x+sa / zsa
    f16*   x16    = (f16*)take((size_t)MTOK * HD * 2);           // x16 / zsa16
    f16*   qkv16  = (f16*)take((size_t)NB * QROWS * 3 * HD * 2); // padded
    f16*   sa16   = (f16*)take((size_t)MTOK * HD * 2);           // sa / mlp hidden

    cvt_f32_f16_kernel<<<(3 * HD * HD + 255) / 256, 256, 0, stream>>>(w_qkv, wqkv16, 3 * HD * HD);
    cvt_f32_f16_kernel<<<(HD * HD + 255) / 256, 256, 0, stream>>>(w_o,  wo16,  HD * HD);
    cvt_f32_f16_kernel<<<(HD * HD + 255) / 256, 256, 0, stream>>>(w_m1, wm116, HD * HD);
    cvt_f32_f16_kernel<<<(HD * HD + 255) / 256, 256, 0, stream>>>(w_m2, wm216, HD * HD);
    pe_kernel<<<(S1T * HD + 255) / 256, 256, 0, stream>>>(petab);
    zero_pad_kernel<<<(NB * (QROWS - S1T) * 3 * HD / 8 + 255) / 256, 256, 0, stream>>>(qkv16);

    embed_ln_kernel<<<MTOK, 256, 0, stream>>>(z, cls, petab, ln_g, ln_b, x32, x16);

    gemm_f16_kernel<0><<<dim3(3 * HD / 64, MTOK / 64), 256, 0, stream>>>(
        x16, wqkv16, b_qkv, nullptr, qkv16, nullptr, MTOK, 3 * HD, HD);

    attn_kernel<<<dim3(10, NB), 320, 0, stream>>>(qkv16, sa16);

    gemm_f16_kernel<1><<<dim3(HD / 64, MTOK / 64), 256, 0, stream>>>(
        sa16, wo16, b_o, x32, nullptr, x32, MTOK, HD, HD);      // x32 <- x + sa

    ln_kernel<<<MTOK, 256, 0, stream>>>(x32, ln_g, ln_b, x32, x16);  // zsa

    gemm_f16_kernel<2><<<dim3(HD / 64, MTOK / 64), 256, 0, stream>>>(
        x16, wm116, b_m1, nullptr, sa16, nullptr, MTOK, HD, HD); // gelu hidden

    gemm_f16_kernel<3><<<dim3(HD / 64, MTOK / 64), 256, 0, stream>>>(
        sa16, wm216, b_m2, x32, nullptr, out, MTOK, HD, HD);     // zsa+mlp -> (S1,B,H)
}
